// MultiHeadedAttention_36885179138580
// MI455X (gfx1250) — matrix operand order repaired and checked by reading, not compiled, not run on hardware
//
#include <hip/hip_runtime.h>
#include <hip/hip_bf16.h>

// MHA for MI455X (gfx1250, wave32). All GEMMs via v_wmma_f32_16x16x32_bf16.
// B=4 T=2048 E=1024 H=16 D=64.  ~137 GFLOP vs ~80MB minimum traffic -> compute
// bound, so everything routes through the bf16 WMMA pipe with f32 accumulation.
// Attention streams 64 keys/iter so the softmax VALU/shfl tail stays hidden
// under the 16 WMMAs per iteration (XDL WMMA co-executes with VALU).

#define BB 4
#define TT 2048
#define EE 1024
#define HH 16
#define DD 64
#define CC (HH * DD)   // 1024

typedef __bf16 bf16_t;
typedef __attribute__((ext_vector_type(16))) __bf16 bf16x16;
typedef __attribute__((ext_vector_type(8)))  float  f32x8;

union FragB16 { bf16x16 v; uint4 u[2]; };

__device__ __forceinline__ f32x8 wmma_bf16(bf16x16 a, bf16x16 b, f32x8 c) {
  // (neg_a, A, neg_b, B, c_mod, C, reuse_a, reuse_b)
  return __builtin_amdgcn_wmma_f32_16x16x32_bf16(false, a, false, b, (short)0, c,
                                                 false, false);
}

__device__ __forceinline__ f32x8 zero8() {
  f32x8 z = {0.f, 0.f, 0.f, 0.f, 0.f, 0.f, 0.f, 0.f};
  return z;
}

// A-fragment (16x32 bf16): lane holds row M=lane%16; lanes 0-15: K=0..15;
// lanes 16-31: K=16..31  ->  two 16B loads at +khalf*16 and +khalf*16+8.
__device__ __forceinline__ bf16x16 load_a(const bf16_t* row, int khalf) {
  FragB16 f;
  f.u[0] = *(const uint4*)(row + khalf * 16);
  f.u[1] = *(const uint4*)(row + khalf * 16 + 8);
  return f.v;
}

// B-fragment (32x16 bf16): lane holds col N=lane%16; lanes 0-15: K=0..15,
// lanes 16-31: K=16..31 -> one contiguous 32B region (ptr already includes khalf*16).
__device__ __forceinline__ bf16x16 load_b(const bf16_t* p) {
  FragB16 f;
  f.u[0] = *(const uint4*)(p);
  f.u[1] = *(const uint4*)(p + 8);
  return f.v;
}

// ---------------- conversion / transpose kernels ----------------

__global__ void k_f32_to_bf16(const float* __restrict__ s, bf16_t* __restrict__ d, int n) {
  int i = blockIdx.x * blockDim.x + threadIdx.x;
  if (i < n) d[i] = (bf16_t)s[i];
}

// W [H,E,D] f32 -> WT [H,D,E] bf16
__global__ void k_transpose_hw(const float* __restrict__ W, bf16_t* __restrict__ WT) {
  int i = blockIdx.x * blockDim.x + threadIdx.x;   // i = (h*D + d)*E + e
  int e = i % EE;
  int r = i / EE;
  int d = r % DD;
  int h = r / DD;
  WT[i] = (bf16_t)W[((size_t)h * EE + e) * DD + d];
}

// Wp [C,E] f32 -> WpT [E,C] bf16
__global__ void k_transpose_wp(const float* __restrict__ Wp, bf16_t* __restrict__ WpT) {
  int i = blockIdx.x * blockDim.x + threadIdx.x;   // i = e*C + c
  int c = i % CC;
  int e = i / CC;
  WpT[i] = (bf16_t)Wp[(size_t)c * EE + e];
}

// ---------------- QKV projection: X[b] (TxE) x W[h]^T -> Q,K bf16 [bh,T,D]; V^T [bh,D,T]

__global__ __launch_bounds__(256) void k_qkv(const bf16_t* __restrict__ xb,
                                             const bf16_t* __restrict__ WqT,
                                             const bf16_t* __restrict__ WkT,
                                             const bf16_t* __restrict__ WvT,
                                             bf16_t* __restrict__ Qb,
                                             bf16_t* __restrict__ Kb,
                                             bf16_t* __restrict__ Vt) {
  const int bh = blockIdx.x;
  const int b = bh / HH, h = bh % HH;
  const int wid   = threadIdx.x >> 5;
  const int lane  = threadIdx.x & 31;
  const int lmod  = lane & 15;
  const int khalf = lane >> 4;
  const int t0 = blockIdx.y * 128 + wid * 16;

  f32x8 aq[4], ak[4], av[4];
#pragma unroll
  for (int n = 0; n < 4; n++) { aq[n] = zero8(); ak[n] = zero8(); av[n] = zero8(); }

  const bf16_t* xrow = xb + ((size_t)b * TT + (t0 + lmod)) * EE;

  for (int ke = 0; ke < EE; ke += 32) {
    bf16x16 a = load_a(xrow + ke, khalf);
#pragma unroll
    for (int n = 0; n < 4; n++) {
      const size_t woff = ((size_t)h * DD + n * 16 + lmod) * EE + ke + khalf * 16;
      aq[n] = wmma_bf16(a, load_b(WqT + woff), aq[n]);
      ak[n] = wmma_bf16(a, load_b(WkT + woff), ak[n]);
      av[n] = wmma_bf16(a, load_b(WvT + woff), av[n]);
    }
  }

  // C/D layout: lane col = n*16+lmod, row = j + 8*khalf
  const size_t base = (size_t)bh * TT * DD;
#pragma unroll
  for (int n = 0; n < 4; n++) {
#pragma unroll
    for (int j = 0; j < 8; j++) {
      const int t = t0 + j + 8 * khalf;
      const int d = n * 16 + lmod;
      Qb[base + (size_t)t * DD + d] = (bf16_t)aq[n][j];
      Kb[base + (size_t)t * DD + d] = (bf16_t)ak[n][j];
      Vt[base + (size_t)d * TT + t] = (bf16_t)av[n][j];   // V stored transposed
    }
  }
}

// ---------------- flash attention: per-wave 16 query rows, stream s in chunks of 64

__global__ __launch_bounds__(256) void k_attn(const bf16_t* __restrict__ Qb,
                                              const bf16_t* __restrict__ Kb,
                                              const bf16_t* __restrict__ Vt,
                                              bf16_t* __restrict__ cat) {
  __shared__ alignas(64) bf16_t ldsP[8][16][64];   // per-wave P-transpose tile (16 KB)

  const int bh = blockIdx.x;
  const int b = bh / HH, h = bh % HH;
  const int wid   = threadIdx.x >> 5;
  const int lane  = threadIdx.x & 31;
  const int lmod  = lane & 15;
  const int khalf = lane >> 4;
  const int t0 = blockIdx.y * 128 + wid * 16;
  const size_t base = (size_t)bh * TT * DD;

  // Q fragments, resident for whole loop (K-dim = D = 64 -> 2 fragments)
  bf16x16 qf[2];
  {
    const bf16_t* qrow = Qb + base + (size_t)(t0 + lmod) * DD;
    qf[0] = load_a(qrow, khalf);
    qf[1] = load_a(qrow + 32, khalf);
  }

  f32x8 O[4];
#pragma unroll
  for (int n = 0; n < 4; n++) O[n] = zero8();
  float m[8], l[8];
#pragma unroll
  for (int j = 0; j < 8; j++) { m[j] = -3.0e38f; l[j] = 0.f; }

  // softmax in exp2 domain: fold 1/sqrt(D) * log2(e) into the score scale
  const float cs = 0.125f * 1.44269504088896340736f;

  for (int s0 = 0; s0 < TT; s0 += 64) {
    if (s0 + 64 < TT) {
      __builtin_prefetch(Kb + base + (size_t)(s0 + 64 + lmod) * DD, 0, 0);
      __builtin_prefetch(Vt + base + (size_t)lmod * TT + s0 + 64, 0, 0);
    }

    // S = Q x K^T : four 16x16 score tiles (s0 .. s0+63)
    f32x8 sc[4];
#pragma unroll
    for (int tile = 0; tile < 4; tile++) {
      const bf16_t* krow = Kb + base + (size_t)(s0 + tile * 16 + lmod) * DD + khalf * 16;
      sc[tile] = wmma_bf16(qf[0], load_b(krow), zero8());
      sc[tile] = wmma_bf16(qf[1], load_b(krow + 32), sc[tile]);
    }

    // online softmax per row j (row lives in lanes khalf*16..khalf*16+15)
#pragma unroll
    for (int j = 0; j < 8; j++) {
      float v0 = sc[0][j] * cs;
      float v1 = sc[1][j] * cs;
      float v2 = sc[2][j] * cs;
      float v3 = sc[3][j] * cs;
      float mx = fmaxf(fmaxf(v0, v1), fmaxf(v2, v3));
      for (int off = 1; off < 16; off <<= 1)
        mx = fmaxf(mx, __shfl_xor(mx, off, 32));
      const float mnew  = fmaxf(m[j], mx);
      const float alpha = exp2f(m[j] - mnew);
      const float p0 = exp2f(v0 - mnew);
      const float p1 = exp2f(v1 - mnew);
      const float p2 = exp2f(v2 - mnew);
      const float p3 = exp2f(v3 - mnew);
      float rs = (p0 + p1) + (p2 + p3);
      for (int off = 1; off < 16; off <<= 1)
        rs += __shfl_xor(rs, off, 32);
      l[j] = l[j] * alpha + rs;
      m[j] = mnew;
#pragma unroll
      for (int n = 0; n < 4; n++) O[n][j] *= alpha;
      // stash P (C layout: row = j+8*khalf, col = tile*16 + lmod)
      ldsP[wid][j + 8 * khalf][lmod]      = (bf16_t)p0;
      ldsP[wid][j + 8 * khalf][16 + lmod] = (bf16_t)p1;
      ldsP[wid][j + 8 * khalf][32 + lmod] = (bf16_t)p2;
      ldsP[wid][j + 8 * khalf][48 + lmod] = (bf16_t)p3;
    }

    // cross-lane LDS RAW within the wave: DS ops are in-order per wave, but
    // make the dependency explicit before the transposed read-back.
    asm volatile("s_wait_dscnt 0" ::: "memory");

    // read P back in A-fragment layout (row = lmod), K = 64 -> 2 fragments
    FragB16 pa0, pa1;
    pa0.u[0] = *(const uint4*)(&ldsP[wid][lmod][khalf * 16]);
    pa0.u[1] = *(const uint4*)(&ldsP[wid][lmod][khalf * 16 + 8]);
    pa1.u[0] = *(const uint4*)(&ldsP[wid][lmod][32 + khalf * 16]);
    pa1.u[1] = *(const uint4*)(&ldsP[wid][lmod][32 + khalf * 16 + 8]);

    // O += P x V   (B-frag from V^T rows: contiguous along s)
#pragma unroll
    for (int n = 0; n < 4; n++) {
      const bf16_t* vrow = Vt + base + (size_t)(n * 16 + lmod) * TT + s0 + khalf * 16;
      O[n] = wmma_bf16(pa0.v, load_b(vrow), O[n]);
      O[n] = wmma_bf16(pa1.v, load_b(vrow + 32), O[n]);
    }
  }

  // normalize and emit concat-layout bf16: cat[b, t, h*D + d]
#pragma unroll
  for (int j = 0; j < 8; j++) {
    const float inv = 1.0f / l[j];
    const int t = t0 + j + 8 * khalf;
#pragma unroll
    for (int n = 0; n < 4; n++) {
      const int c = h * DD + n * 16 + lmod;
      cat[((size_t)b * TT + t) * CC + c] = (bf16_t)(O[n][j] * inv);
    }
  }
}

// ---------------- output projection: cat (B*T x C) x Wp (C x E) + bp -> f32

__global__ __launch_bounds__(256) void k_proj(const bf16_t* __restrict__ cat,
                                              const bf16_t* __restrict__ WpT,
                                              const float* __restrict__ bp,
                                              float* __restrict__ out) {
  const int wid   = threadIdx.x >> 5;
  const int lane  = threadIdx.x & 31;
  const int lmod  = lane & 15;
  const int khalf = lane >> 4;
  const int r0 = blockIdx.x * 128 + wid * 16;   // row over B*T
  const int e0 = blockIdx.y * 64;

  f32x8 acc[4];
#pragma unroll
  for (int n = 0; n < 4; n++) acc[n] = zero8();

  const bf16_t* crow = cat + (size_t)(r0 + lmod) * CC;
  for (int kc = 0; kc < CC; kc += 32) {
    bf16x16 a = load_a(crow + kc, khalf);
#pragma unroll
    for (int n = 0; n < 4; n++) {
      const bf16_t* wrow = WpT + (size_t)(e0 + n * 16 + lmod) * CC + kc + khalf * 16;
      acc[n] = wmma_bf16(a, load_b(wrow), acc[n]);
    }
  }

#pragma unroll
  for (int n = 0; n < 4; n++) {
#pragma unroll
    for (int j = 0; j < 8; j++) {
      const int r = r0 + j + 8 * khalf;
      const int e = e0 + n * 16 + lmod;
      out[(size_t)r * EE + e] = acc[n][j] + bp[e];
    }
  }
}

// ---------------- launch ----------------

extern "C" void kernel_launch(void* const* d_in, const int* in_sizes, int n_in,
                              void* d_out, int out_size, void* d_ws, size_t ws_size,
                              hipStream_t stream) {
  const float* x  = (const float*)d_in[0];
  const float* Wq = (const float*)d_in[1];
  const float* Wk = (const float*)d_in[2];
  const float* Wv = (const float*)d_in[3];
  const float* Wp = (const float*)d_in[4];
  const float* bp = (const float*)d_in[5];
  float* out = (float*)d_out;

  // workspace carve-up (bf16 elements) — total ≈ 92 MB
  bf16_t* ws = (bf16_t*)d_ws;
  size_t off = 0;
  bf16_t* xb  = ws + off; off += (size_t)BB * TT * EE;   // 8M
  bf16_t* WqT = ws + off; off += (size_t)HH * EE * DD;   // 1M
  bf16_t* WkT = ws + off; off += (size_t)HH * EE * DD;
  bf16_t* WvT = ws + off; off += (size_t)HH * EE * DD;
  bf16_t* WpT = ws + off; off += (size_t)CC * EE;        // 1M
  bf16_t* Qb  = ws + off; off += (size_t)BB * HH * TT * DD;  // 8M
  bf16_t* Kb  = ws + off; off += (size_t)BB * HH * TT * DD;
  bf16_t* Vt  = ws + off; off += (size_t)BB * HH * TT * DD;
  bf16_t* cat = ws + off; off += (size_t)BB * TT * CC;   // 8M

  const int nX  = BB * TT * EE;     // 8,388,608
  const int nW  = HH * EE * DD;     // 1,048,576
  const int nWp = CC * EE;          // 1,048,576

  k_f32_to_bf16<<<(nX + 255) / 256, 256, 0, stream>>>(x, xb, nX);
  k_transpose_hw<<<(nW + 255) / 256, 256, 0, stream>>>(Wq, WqT);
  k_transpose_hw<<<(nW + 255) / 256, 256, 0, stream>>>(Wk, WkT);
  k_transpose_hw<<<(nW + 255) / 256, 256, 0, stream>>>(Wv, WvT);
  k_transpose_wp<<<(nWp + 255) / 256, 256, 0, stream>>>(Wp, WpT);

  k_qkv<<<dim3(BB * HH, TT / 128), 256, 0, stream>>>(xb, WqT, WkT, WvT, Qb, Kb, Vt);
  k_attn<<<dim3(BB * HH, TT / 128), 256, 0, stream>>>(Qb, Kb, Vt, cat);
  k_proj<<<dim3(BB * TT / 128, EE / 64), 256, 0, stream>>>(cat, WpT, bp, out);
}
